// FixedAttentionTransformerBlock_86835648790968
// MI455X (gfx1250) — matrix operand order repaired and checked by reading, not compiled, not run on hardware
//
#include <hip/hip_runtime.h>
#include <hip/hip_bf16.h>
#include <stdint.h>

#define B_  8
#define L_  1024
#define E_  256
#define H_  8
#define DH_ 32

typedef __bf16 bf16;
typedef __attribute__((ext_vector_type(16))) __bf16 v16bf;
typedef __attribute__((ext_vector_type(8)))  float  v8f;
typedef __attribute__((ext_vector_type(4)))  unsigned int v4u;
typedef __attribute__((ext_vector_type(8)))  unsigned int v8u;

union FragAB { v16bf v; uint4 q[2]; };

static __device__ __forceinline__ bf16 f2bf(float f) { return (bf16)f; }

// ---------------------------------------------------------------- TDM 2D tile load: global (row-major, 2B elems) -> LDS
// D# per CDNA5 ISA ch.8: group0 = {count/flags, lds_addr, global_addr, type=2},
// group1 = {data_size=2B, tensor_dim0/1, tile_dim0/1, tensor_dim0_stride}.
static __device__ __forceinline__ void tdm_load_2d(const void* gptr, unsigned int lds_off,
                                                   unsigned int tdim0, unsigned int tdim1,
                                                   unsigned int stride0,
                                                   unsigned int tile0, unsigned int tile1) {
  unsigned long long ga = (unsigned long long)gptr;
  v4u g0;
  g0[0] = 1u;                                              // count=1, user descriptor
  g0[1] = lds_off;                                         // lds_addr (bytes)
  g0[2] = (unsigned int)ga;                                // global_addr[31:0]
  g0[3] = (unsigned int)((ga >> 32) & 0x1FFFFFFu) | (2u << 30); // addr[56:32] | type=2
  v8u g1;
  g1[0] = 1u << 16;                                        // data_size = 2 bytes
  g1[1] = (tdim0 & 0xFFFFu) << 16;                         // tensor_dim0[15:0]
  g1[2] = (tdim0 >> 16) | ((tdim1 & 0xFFFFu) << 16);       // tensor_dim0[31:16] | tensor_dim1[15:0]
  g1[3] = (tdim1 >> 16) | (tile0 << 16);                   // tensor_dim1[31:16] | tile_dim0
  g1[4] = tile1;                                           // tile_dim1 (tile_dim2 = 0)
  g1[5] = stride0;                                         // tensor_dim0_stride[31:0]
  g1[6] = 0u;                                              // stride hi / tensor_dim1_stride lo
  g1[7] = 0u;
  asm volatile("tensor_load_to_lds %0, %1" :: "s"(g0), "s"(g1) : "memory");
}

// ---------------------------------------------------------------- f32 (KxN) -> bf16 transposed (NxK)
__global__ void cvt_transpose_bf16(const float* __restrict__ in, bf16* __restrict__ out,
                                   int K, int N) {
  int k = blockIdx.x * 16 + (threadIdx.x & 15);
  int n = blockIdx.y * 16 + (threadIdx.x >> 4);
  out[(size_t)n * K + k] = f2bf(in[(size_t)k * N + n]);
}

// ---------------------------------------------------------------- LayerNorm (optionally fused residual)
__global__ void ln_kernel(const float* __restrict__ x, const float* __restrict__ res,
                          const float* __restrict__ g, const float* __restrict__ bta,
                          float* __restrict__ outF, bf16* __restrict__ outB) {
  int row = blockIdx.x;
  int tid = threadIdx.x;
  float v = x[(size_t)row * E_ + tid];
  if (res) v += res[(size_t)row * E_ + tid];
  float s = v, s2 = v * v;
  #pragma unroll
  for (int o = 16; o >= 1; o >>= 1) {
    s  += __shfl_xor(s,  o, 32);
    s2 += __shfl_xor(s2, o, 32);
  }
  __shared__ float ps[8], ps2[8];
  int wid = tid >> 5, lane = tid & 31;
  if (lane == 0) { ps[wid] = s; ps2[wid] = s2; }
  __syncthreads();
  float ts = 0.f, ts2 = 0.f;
  #pragma unroll
  for (int i = 0; i < 8; i++) { ts += ps[i]; ts2 += ps2[i]; }
  float mu   = ts  * (1.0f / E_);
  float var  = ts2 * (1.0f / E_) - mu * mu;
  float rstd = rsqrtf(var + 1e-5f);
  float y = (v - mu) * rstd * g[tid] + bta[tid];
  outF[(size_t)row * E_ + tid] = y;
  outB[(size_t)row * E_ + tid] = f2bf(y);
}

// ---------------------------------------------------------------- softmax row stats (online m,s per causal row)
__global__ void rowstats_kernel(const float* __restrict__ d, const float* __restrict__ gamma,
                                float* __restrict__ rmax, float* __restrict__ rsum) {
  int wid  = blockIdx.x * 8 + (threadIdx.x >> 5);
  int lane = threadIdx.x & 31;
  int i  = wid & (L_ - 1);
  int bh = wid >> 10;
  int h  = bh & (H_ - 1);
  int b  = bh >> 3;
  float g = gamma[h];
  const float* drow = d + ((size_t)b * L_ + i) * L_;
  float m = -3.0e38f, s = 0.0f;
  for (int j = lane; j <= i; j += 32) {
    float dj = drow[j];
    float w  = -g * dj * dj;
    float nm = fmaxf(m, w);
    s = s * __expf(m - nm) + __expf(w - nm);
    m = nm;
  }
  #pragma unroll
  for (int o = 16; o >= 1; o >>= 1) {
    float om = __shfl_xor(m, o, 32);
    float os = __shfl_xor(s, o, 32);
    float nm = fmaxf(m, om);
    s = s * __expf(m - nm) + os * __expf(om - nm);
    m = nm;
  }
  if (lane == 0) { rmax[wid] = m; rsum[wid] = s; }
}

// ---------------------------------------------------------------- attention: ctx = softmax(-g d^2, causal) @ V
// V stored transposed: vt[(b*H+h)][dh][j] so the V tile is a plain 2D TDM copy -> Vs[dh][k]
__global__ void attn_kernel(const float* __restrict__ d, const float* __restrict__ gamma,
                            const float* __restrict__ rmax, const float* __restrict__ rsum,
                            const bf16* __restrict__ vt, bf16* __restrict__ ctx) {
  __shared__ alignas(16) bf16 Vs[DH_][32];   // [dh][k], k = j - j0
  int blk     = blockIdx.x;
  int tileblk = blk & 7;                     // L/128 = 8
  int bh      = blk >> 3;
  int h = bh & 7, b = bh >> 3;
  int i0blk = tileblk * 128;
  int tid = threadIdx.x;
  int w = tid >> 5, lane = tid & 31;
  int row = lane & 15, half = lane >> 4;
  int i0 = i0blk + w * 16;
  int i  = i0 + row;
  float g     = gamma[h];
  float m_i   = rmax[(size_t)bh * L_ + i];
  float inv_s = 1.0f / rsum[(size_t)bh * L_ + i];
  const float* drow = d + ((size_t)b * L_ + i) * L_;
  bool issuer = (__builtin_amdgcn_readfirstlane(tid) == 0);      // wave 0, scalar branch
  unsigned int vsbase = (unsigned int)(size_t)(void*)&Vs[0][0];
  v8f acc0 = {}; v8f acc1 = {};
  int numJ = (i0blk + 127) / 32 + 1;
  for (int t = 0; t < numJ; ++t) {
    int j0 = t * 32;
    if (issuer) {
      tdm_load_2d(vt + (size_t)bh * DH_ * L_ + j0, vsbase,
                  L_, DH_, L_, 32, DH_);
      __builtin_amdgcn_s_wait_tensorcnt(0);
    }
    __syncthreads();
    if (__builtin_amdgcn_readfirstlane((int)(j0 <= i0 + 15))) {  // wave-uniform, EXEC untouched
      union { float4 f4[4]; float f[16]; } du;
      du.f4[0] = *(const float4*)(drow + j0 + 8 * half);
      du.f4[1] = *(const float4*)(drow + j0 + 8 * half + 4);
      du.f4[2] = *(const float4*)(drow + j0 + 16 + 8 * half);
      du.f4[3] = *(const float4*)(drow + j0 + 16 + 8 * half + 4);
      FragAB pf;
      #pragma unroll
      for (int e = 0; e < 16; e++) {
        int j = j0 + ((e < 8) ? (8 * half + e) : (16 + 8 * half + (e - 8)));
        float dj  = du.f[e];
        float arg = fmaf(-g * dj, dj, -m_i);
        arg = (j <= i) ? arg : -1.0e30f;                         // branchless causal mask
        pf.v[e] = f2bf(__expf(arg) * inv_s);
      }
      FragAB vf0, vf1;
      vf0.q[0] = *(const uint4*)&Vs[row][8 * half];
      vf0.q[1] = *(const uint4*)&Vs[row][16 + 8 * half];
      vf1.q[0] = *(const uint4*)&Vs[16 + row][8 * half];
      vf1.q[1] = *(const uint4*)&Vs[16 + row][16 + 8 * half];
      acc0 = __builtin_amdgcn_wmma_f32_16x16x32_bf16(false, pf.v, false, vf0.v, (short)0, acc0, false, false);
      acc1 = __builtin_amdgcn_wmma_f32_16x16x32_bf16(false, pf.v, false, vf1.v, (short)0, acc1, false, false);
    }
    __syncthreads();
  }
  #pragma unroll
  for (int r = 0; r < 8; r++) {
    int iout = i0 + r + 8 * half;
    int col  = lane & 15;
    bf16* dst = ctx + ((size_t)b * L_ + iout) * E_ + h * DH_;
    dst[col]      = f2bf(acc0[r]);
    dst[16 + col] = f2bf(acc1[r]);
  }
}

// ---------------------------------------------------------------- tiled bf16 WMMA GEMM, TDM-fed, double-buffered
// A: MxK row-major bf16; Wt: NxK row-major bf16 (pre-transposed weights)
#define BM 64
#define BN 128
#define BK 32
__global__ void gemm_bf16_kernel(const bf16* __restrict__ A, const bf16* __restrict__ Wt,
                                 const float* __restrict__ bias,
                                 float* __restrict__ outF, bf16* __restrict__ outB,
                                 int M, int N, int K, int relu, int vtrans) {
  __shared__ alignas(16) bf16 As[2][BM][BK];
  __shared__ alignas(16) bf16 Bs[2][BN][BK];   // [n][k]
  int nt = blockIdx.x, mt = blockIdx.y;
  int m0 = mt * BM, n0 = nt * BN;
  int tid = threadIdx.x;
  int w = tid >> 5, lane = tid & 31;
  int half = lane >> 4, lr = lane & 15;
  bool issuer = (__builtin_amdgcn_readfirstlane(tid) == 0);
  unsigned int asbase = (unsigned int)(size_t)(void*)&As[0][0][0];
  unsigned int bsbase = (unsigned int)(size_t)(void*)&Bs[0][0][0];
  v8f acc[4] = {};
  int nK = K / BK;
  if (issuer) {   // prologue: DMA first K-tile of A and B into buffer 0
    tdm_load_2d(A  + (size_t)m0 * K, asbase, (unsigned)K, (unsigned)M, (unsigned)K, BK, BM);
    tdm_load_2d(Wt + (size_t)n0 * K, bsbase, (unsigned)K, (unsigned)N, (unsigned)K, BK, BN);
    __builtin_amdgcn_s_wait_tensorcnt(0);
  }
  __syncthreads();
  for (int kt = 0; kt < nK; ++kt) {
    int cur = kt & 1, nxt = cur ^ 1;
    if ((kt + 1 < nK) && issuer) {   // prefetch next K-tile via TDM while computing
      int k0n = (kt + 1) * BK;
      tdm_load_2d(A  + (size_t)m0 * K + k0n, asbase + (unsigned)(nxt * BM * BK * 2),
                  (unsigned)K, (unsigned)M, (unsigned)K, BK, BM);
      tdm_load_2d(Wt + (size_t)n0 * K + k0n, bsbase + (unsigned)(nxt * BN * BK * 2),
                  (unsigned)K, (unsigned)N, (unsigned)K, BK, BN);
    }
    FragAB bfr;
    bfr.q[0] = *(const uint4*)&Bs[cur][w * 16 + lr][8 * half];
    bfr.q[1] = *(const uint4*)&Bs[cur][w * 16 + lr][16 + 8 * half];
    #pragma unroll
    for (int s = 0; s < 4; s++) {
      FragAB af;
      af.q[0] = *(const uint4*)&As[cur][s * 16 + lr][8 * half];
      af.q[1] = *(const uint4*)&As[cur][s * 16 + lr][16 + 8 * half];
      acc[s] = __builtin_amdgcn_wmma_f32_16x16x32_bf16(false, af.v, false, bfr.v, (short)0, acc[s], false, false);
    }
    if (issuer) __builtin_amdgcn_s_wait_tensorcnt(0);
    __syncthreads();
  }
  #pragma unroll
  for (int s = 0; s < 4; s++) {
    #pragma unroll
    for (int r = 0; r < 8; r++) {
      int gm = m0 + s * 16 + r + 8 * half;
      int gn = n0 + w * 16 + lr;
      float v = acc[s][r] + bias[gn];
      if (relu) v = fmaxf(v, 0.0f);
      if (outF) outF[(size_t)gm * N + gn] = v;
      if (outB) {
        if (vtrans) {  // write V transposed: vt[(b*H+h)][dh][l]
          int b = gm >> 10, l = gm & (L_ - 1);
          int hh = gn >> 5, dh = gn & 31;
          outB[(((size_t)(b * H_ + hh)) * DH_ + dh) * L_ + l] = f2bf(v);
        } else {
          outB[(size_t)gm * N + gn] = f2bf(v);
        }
      }
    }
  }
}

// ---------------------------------------------------------------- out = e2 + (padding ? 0 : e3)
__global__ void final_kernel(const float* __restrict__ e2, const float* __restrict__ e3,
                             float* __restrict__ out, int n) {
  int idx = blockIdx.x * 256 + threadIdx.x;
  if (idx < n) {
    int l = (idx >> 8) & (L_ - 1);
    float v = e2[idx];
    if (l < L_ - 128) v += e3[idx];
    out[idx] = v;
  }
}

// ---------------------------------------------------------------- launcher
extern "C" void kernel_launch(void* const* d_in, const int* in_sizes, int n_in,
                              void* d_out, int out_size, void* d_ws, size_t ws_size,
                              hipStream_t stream) {
  // input order: tokens, e, d, causal_mask, padding_mask, ln1_g, ln1_b, ln2_g, ln2_b,
  //              Wv, bv, Wo, bo, W1, b1, W2, b2, gamma
  const float* e     = (const float*)d_in[1];
  const float* d_    = (const float*)d_in[2];
  const float* ln1_g = (const float*)d_in[5];
  const float* ln1_b = (const float*)d_in[6];
  const float* ln2_g = (const float*)d_in[7];
  const float* ln2_b = (const float*)d_in[8];
  const float* Wv    = (const float*)d_in[9];
  const float* bv    = (const float*)d_in[10];
  const float* Wo    = (const float*)d_in[11];
  const float* bo    = (const float*)d_in[12];
  const float* W1    = (const float*)d_in[13];
  const float* b1    = (const float*)d_in[14];
  const float* W2    = (const float*)d_in[15];
  const float* b2    = (const float*)d_in[16];
  const float* gamma = (const float*)d_in[17];
  float* out = (float*)d_out;

  char* ws = (char*)d_ws;
  size_t off = 0;
  auto alloc = [&](size_t bytes) { char* p = ws + off; off = (off + bytes + 255) & ~(size_t)255; return p; };
  float* e0f   = (float*)alloc((size_t)B_ * L_ * E_ * 4);
  bf16*  e0b   = (bf16*) alloc((size_t)B_ * L_ * E_ * 2);
  bf16*  WvT   = (bf16*) alloc((size_t)E_ * E_ * 2);
  bf16*  WoT   = (bf16*) alloc((size_t)E_ * E_ * 2);
  bf16*  W1T   = (bf16*) alloc((size_t)E_ * 4 * E_ * 2);
  bf16*  W2T   = (bf16*) alloc((size_t)4 * E_ * E_ * 2);
  bf16*  vt    = (bf16*) alloc((size_t)B_ * H_ * DH_ * L_ * 2);
  float* rmax  = (float*)alloc((size_t)B_ * H_ * L_ * 4);
  float* rsum  = (float*)alloc((size_t)B_ * H_ * L_ * 4);
  bf16*  ctxb  = (bf16*) alloc((size_t)B_ * L_ * E_ * 2);
  float* e1f   = (float*)alloc((size_t)B_ * L_ * E_ * 4);
  float* e2f   = (float*)alloc((size_t)B_ * L_ * E_ * 4);
  bf16*  e2b   = (bf16*) alloc((size_t)B_ * L_ * E_ * 2);
  bf16*  hb    = (bf16*) alloc((size_t)B_ * L_ * 4 * E_ * 2);
  float* e3f   = (float*)alloc((size_t)B_ * L_ * E_ * 4);
  (void)ws_size; (void)in_sizes; (void)n_in; (void)out_size;

  const int rows = B_ * L_;               // 8192
  // transposed bf16 weight conversions (Wt is NxK)
  cvt_transpose_bf16<<<dim3(E_ / 16, E_ / 16), 256, 0, stream>>>(Wv, WvT, E_, E_);
  cvt_transpose_bf16<<<dim3(E_ / 16, E_ / 16), 256, 0, stream>>>(Wo, WoT, E_, E_);
  cvt_transpose_bf16<<<dim3(E_ / 16, 4 * E_ / 16), 256, 0, stream>>>(W1, W1T, E_, 4 * E_);
  cvt_transpose_bf16<<<dim3(4 * E_ / 16, E_ / 16), 256, 0, stream>>>(W2, W2T, 4 * E_, E_);
  // e0 = LN1(e)
  ln_kernel<<<rows, 256, 0, stream>>>(e, nullptr, ln1_g, ln1_b, e0f, e0b);
  // v = e0 @ Wv + bv, written transposed per head -> vt
  gemm_bf16_kernel<<<dim3(E_ / BN, rows / BM), 256, 0, stream>>>(e0b, WvT, bv, nullptr, vt, rows, E_, E_, 0, 1);
  // softmax row stats over causal radial scores
  rowstats_kernel<<<(B_ * H_ * L_) / 8, 256, 0, stream>>>(d_, gamma, rmax, rsum);
  // ctx = attn @ v
  attn_kernel<<<(B_ * H_) * (L_ / 128), 256, 0, stream>>>(d_, gamma, rmax, rsum, vt, ctxb);
  // e1 = ctx @ Wo + bo
  gemm_bf16_kernel<<<dim3(E_ / BN, rows / BM), 256, 0, stream>>>(ctxb, WoT, bo, e1f, nullptr, rows, E_, E_, 0, 0);
  // e2 = LN2(e1 + e0)
  ln_kernel<<<rows, 256, 0, stream>>>(e1f, e0f, ln2_g, ln2_b, e2f, e2b);
  // h = relu(e2 @ W1 + b1)
  gemm_bf16_kernel<<<dim3(4 * E_ / BN, rows / BM), 256, 0, stream>>>(e2b, W1T, b1, nullptr, hb, rows, 4 * E_, E_, 1, 0);
  // e3 = h @ W2 + b2
  gemm_bf16_kernel<<<dim3(E_ / BN, rows / BM), 256, 0, stream>>>(hb, W2T, b2, e3f, nullptr, rows, E_, 4 * E_, 0, 0);
  // out = e2 + masked(e3)
  final_kernel<<<(B_ * L_ * E_) / 256, 256, 0, stream>>>(e2f, e3f, out, B_ * L_ * E_);
}
